// CausalSelfAttention_15573551415431
// MI455X (gfx1250) — matrix operand order repaired and checked by reading, not compiled, not run on hardware
//
#include <hip/hip_runtime.h>
#include <stdint.h>

// ---------------- problem constants ----------------
#define B_  2
#define T_  2048
#define E_  2048
#define H_  32
#define G_  8
#define D_  64
#define S_  2048
#define QKV_F ((H_ + 2*G_) * D_)          // 3072
#define NEGV (-0.7f * 3.402823466e+38f)

typedef __attribute__((ext_vector_type(16))) __bf16 v16bf;
typedef __attribute__((ext_vector_type(2)))  __bf16 v2bf;
typedef __attribute__((ext_vector_type(8)))  float  v8f;
typedef int v4i __attribute__((vector_size(16)));   // matches builtin's pointee type

union Frag {
  v16bf    bf;
  uint32_t u[8];
  uint4    q[2];
};

// ---- capability probes ----
#if defined(__has_builtin)
# if __has_builtin(__builtin_amdgcn_cvt_pk_bf16_f32)
#  define HAVE_CVT_PK 1
# endif
# if __has_builtin(__builtin_amdgcn_global_load_async_to_lds_b128)
#  define HAVE_ASYNC_LDS 1
# endif
# if __has_builtin(__builtin_amdgcn_s_wait_asynccnt)
#  define HAVE_WAIT_ASYNC 1
# endif
#endif

#define AS1 __attribute__((address_space(1)))
#define AS3 __attribute__((address_space(3)))

__device__ __forceinline__ uint32_t pack_bf16(float lo, float hi) {
#ifdef HAVE_CVT_PK
  union { v2bf v; uint32_t u; } c;
  c.v = __builtin_amdgcn_cvt_pk_bf16_f32(lo, hi);
  return c.u;
#else
  uint32_t ulo = __float_as_uint(lo);
  uint32_t uhi = __float_as_uint(hi);
  ulo = (ulo + 0x7FFFu + ((ulo >> 16) & 1u)) >> 16;
  uhi = (uhi + 0x7FFFu + ((uhi >> 16) & 1u)) & 0xFFFF0000u;
  return (ulo & 0xFFFFu) | uhi;
#endif
}
__device__ __forceinline__ uint16_t bf16_1(float f) {
  uint32_t u = __float_as_uint(f);
  return (uint16_t)((u + 0x7FFFu + ((u >> 16) & 1u)) >> 16);
}

// async global->LDS copy of one 16B chunk (per-lane), with fallback
__device__ __forceinline__ void copy16_g2l(const void* g, void* l) {
#ifdef HAVE_ASYNC_LDS
  __builtin_amdgcn_global_load_async_to_lds_b128(
      (AS1 v4i*)(uintptr_t)g,
      (AS3 v4i*)(uint32_t)(uintptr_t)l, 0, 0);
#else
  *(uint4*)l = *(const uint4*)g;
#endif
}
__device__ __forceinline__ void wait_async_copies() {
#ifdef HAVE_ASYNC_LDS
# ifdef HAVE_WAIT_ASYNC
  __builtin_amdgcn_s_wait_asynccnt(0);
# else
  asm volatile("s_wait_asynccnt 0" ::: "memory");
# endif
#endif
}

// =====================================================================
// fp32 -> packed bf16 conversion pass (bandwidth-bound, done once)
// =====================================================================
__global__ __launch_bounds__(256)
void convert_f32_bf16(const float* __restrict__ src, uint32_t* __restrict__ dst,
                      int npairs)
{
  const int i = blockIdx.x * 256 + threadIdx.x;
  if (i < npairs) {
    const float2 f = ((const float2*)src)[i];
    dst[i] = pack_bf16(f.x, f.y);
  }
}

// =====================================================================
// C[M,N] = A[M,K] * B[N,K]^T  with bf16 inputs, fp32 out.
// BM=BN=128, BK=32, 256 threads = 8 waves, each wave 64x32 (8 WMMA accs).
// Tile staging is a pure byte copy (async global->LDS when available).
// =====================================================================
#define BM 128
#define BN 128
#define BK 32
#define LDA 20          // LDS row stride in dwords (16 data + 4 pad)

__global__ __launch_bounds__(256)
void gemm_bf16_wmma(const uint16_t* __restrict__ A, const uint16_t* __restrict__ Bm,
                    float* __restrict__ C, int M, int N, int K)
{
  __shared__ uint32_t sA[2][BM * LDA];
  __shared__ uint32_t sB[2][BN * LDA];

  const int tid  = threadIdx.x;
  const int lane = tid & 31;
  const int wave = tid >> 5;
  const int hf   = lane >> 4;
  const int l16  = lane & 15;

  const int bn = blockIdx.x;
  const int bm = blockIdx.y;
  const int wm = wave & 1;
  const int wn = wave >> 1;

  // 512 16B-chunks per tile (128 rows x 4 chunks); 2 chunks/thread/matrix
  auto load_tile = [&](int kt, int buf) {
    const int kb = kt * BK;
    #pragma unroll
    for (int p = 0; p < 2; ++p) {
      const int c   = tid + p * 256;
      const int row = c >> 2;
      const int sub = c & 3;
      copy16_g2l(A + (size_t)(bm * BM + row) * K + kb + sub * 8,
                 &sA[buf][row * LDA + sub * 4]);
      copy16_g2l(Bm + (size_t)(bn * BN + row) * K + kb + sub * 8,
                 &sB[buf][row * LDA + sub * 4]);
    }
  };

  v8f acc[4][2];
  #pragma unroll
  for (int mt = 0; mt < 4; ++mt)
    #pragma unroll
    for (int nt = 0; nt < 2; ++nt) {
      v8f z = {0.f,0.f,0.f,0.f,0.f,0.f,0.f,0.f};
      acc[mt][nt] = z;
    }

  const int NK = K / BK;
  load_tile(0, 0);
  wait_async_copies();
  __syncthreads();

  for (int kt = 0; kt < NK; ++kt) {
    const int buf = kt & 1;
    if (kt + 1 < NK) load_tile(kt + 1, buf ^ 1);

    Frag a[4], bfr[2];
    #pragma unroll
    for (int mt = 0; mt < 4; ++mt) {
      const uint32_t* rp = &sA[buf][(wm * 64 + mt * 16 + l16) * LDA];
      a[mt].q[0] = *(const uint4*)(rp + hf * 8);
      a[mt].q[1] = *(const uint4*)(rp + hf * 8 + 4);
    }
    #pragma unroll
    for (int nt = 0; nt < 2; ++nt) {
      const uint32_t* rp = &sB[buf][(wn * 32 + nt * 16 + l16) * LDA + hf * 8];
      bfr[nt].q[0] = *(const uint4*)(rp);
      bfr[nt].q[1] = *(const uint4*)(rp + 4);
    }
    #pragma unroll
    for (int mt = 0; mt < 4; ++mt)
      #pragma unroll
      for (int nt = 0; nt < 2; ++nt)
        acc[mt][nt] = __builtin_amdgcn_wmma_f32_16x16x32_bf16(
            false, a[mt].bf, false, bfr[nt].bf, (short)0, acc[mt][nt], false, false);

    wait_async_copies();
    __syncthreads();
  }

  #pragma unroll
  for (int mt = 0; mt < 4; ++mt)
    #pragma unroll
    for (int nt = 0; nt < 2; ++nt) {
      const int col = bn * BN + wn * 32 + nt * 16 + l16;
      #pragma unroll
      for (int r = 0; r < 8; ++r) {
        const int row = bm * BM + wm * 64 + mt * 16 + r + hf * 8;
        C[(size_t)row * N + col] = acc[mt][nt][r];
      }
    }
}

// =====================================================================
// Fused RMS-norm + RoPE + cache scatter.
//  - q: bf16, softmax scale folded in -> qb
//  - k: fp32 -> cache_k (output) AND bf16 -> kb shadow
//  - v: fp32 -> cache_v (output) AND bf16 transposed [b,g,d,s] -> vtb shadow
// =====================================================================
__global__ __launch_bounds__(256)
void norm_rope_cache(const float* __restrict__ qkv,
                     const float* __restrict__ qw, const float* __restrict__ kw,
                     const float* __restrict__ cosp, const float* __restrict__ sinp,
                     const int* __restrict__ input_pos,
                     uint16_t* __restrict__ qb, float* __restrict__ ck,
                     float* __restrict__ cv, uint16_t* __restrict__ kb,
                     uint16_t* __restrict__ vtb)
{
  const int bt   = blockIdx.x;
  const int b    = bt / T_;
  const int t    = bt % T_;
  const int lane = threadIdx.x & 31;
  const int wave = threadIdx.x >> 5;

  const float c   = cosp[t * (D_ / 2) + lane];
  const float s   = sinp[t * (D_ / 2) + lane];
  const int   pos = input_pos[t];
  const float* base = qkv + (size_t)bt * QKV_F;

  #pragma unroll
  for (int i = 0; i < 6; ++i) {
    const int head = wave + i * 8;
    float v1 = base[head * D_ + lane];
    float v2 = base[head * D_ + lane + 32];
    if (head < H_ + G_) {
      float ss = v1 * v1 + v2 * v2;
      #pragma unroll
      for (int off = 16; off > 0; off >>= 1) ss += __shfl_xor(ss, off, 32);
      const float rn = rsqrtf(ss * (1.0f / D_) + 1e-6f);
      const float* w = (head < H_) ? qw : kw;
      v1 = v1 * rn * w[lane];
      v2 = v2 * rn * w[lane + 32];
      const float o1 = v1 * c - v2 * s;
      const float o2 = v2 * c + v1 * s;
      if (head < H_) {
        uint16_t* qp = qb + ((size_t)bt * H_ + head) * D_;
        qp[lane]      = bf16_1(o1 * 0.125f);     // fold 1/sqrt(D)
        qp[lane + 32] = bf16_1(o2 * 0.125f);
      } else {
        const int g = head - H_;
        const size_t kidx = ((size_t)(b * S_ + pos) * G_ + g) * D_;
        ck[kidx + lane]      = o1;
        ck[kidx + lane + 32] = o2;
        kb[kidx + lane]      = bf16_1(o1);
        kb[kidx + lane + 32] = bf16_1(o2);
      }
    } else {
      const int g = head - H_ - G_;
      const size_t vidx = ((size_t)(b * S_ + pos) * G_ + g) * D_;
      cv[vidx + lane]      = v1;
      cv[vidx + lane + 32] = v2;
      const size_t tbase = ((size_t)(b * G_ + g)) * D_;
      vtb[(tbase + lane)      * S_ + pos] = bf16_1(v1);
      vtb[(tbase + lane + 32) * S_ + pos] = bf16_1(v2);
    }
  }
}

// =====================================================================
// Flash attention (barrier-free): one block per (b,h,64-row q tile),
// 4 independent waves x 16 query rows.  All WMMA fragments except P are
// loaded directly from global bf16 (K rows and V^T rows are contiguous).
// Only the P C->A relayout round-trips through per-wave LDS.
// =====================================================================
#define NB  32
#define LPA 20          // sP row stride (dwords)

__global__ __launch_bounds__(128)
void attention_kernel(const uint16_t* __restrict__ qb, const uint16_t* __restrict__ kb,
                      const uint16_t* __restrict__ vtb, uint16_t* __restrict__ ob)
{
  __shared__ uint16_t sP16[4][16 * (LPA * 2)];

  const int nq = T_ / 64;
  int idx = blockIdx.x;
  const int qt = idx % nq;  idx /= nq;
  const int h  = idx % H_;  idx /= H_;
  const int b  = idx;
  const int g  = h / (H_ / G_);

  const int lane = threadIdx.x & 31;
  const int wave = threadIdx.x >> 5;
  const int hf   = lane >> 4;
  const int l16  = lane & 15;
  const int qbase = qt * 64 + wave * 16;

  // ---- Q A-fragments (2 K-steps over D=64), direct from global bf16 ----
  Frag qa[2];
  {
    const uint16_t* qp = qb + ((size_t)(b * T_ + qbase + l16) * H_ + h) * D_;
    #pragma unroll
    for (int j = 0; j < 2; ++j) {
      qa[j].q[0] = *(const uint4*)(qp + j * 32 + hf * 16);
      qa[j].q[1] = *(const uint4*)(qp + j * 32 + hf * 16 + 8);
    }
  }

  v8f o[4];
  #pragma unroll
  for (int dt = 0; dt < 4; ++dt) {
    v8f z = {0.f,0.f,0.f,0.f,0.f,0.f,0.f,0.f};
    o[dt] = z;
  }
  float mrow[8], lrow[8];
  #pragma unroll
  for (int r = 0; r < 8; ++r) { mrow[r] = -__builtin_inff(); lrow[r] = 0.0f; }

  const uint16_t* kbase = kb  + ((size_t)b * S_) * (G_ * D_) + (size_t)g * D_;
  const uint16_t* vbase = vtb + ((size_t)(b * G_ + g)) * D_ * S_;

  const int nkb = qt * 2 + 2;              // causal: stop at the diagonal
  for (int kbk = 0; kbk < nkb; ++kbk) {
    // ---- S = Q K^T : 2 n-tiles x 2 K-steps, B-frags straight from global ----
    v8f sfr[2];
    #pragma unroll
    for (int nt = 0; nt < 2; ++nt) {
      v8f sv = {0.f,0.f,0.f,0.f,0.f,0.f,0.f,0.f};
      #pragma unroll
      for (int j = 0; j < 2; ++j) {
        Frag bk;
        const uint16_t* kp =
            kbase + (size_t)(kbk * NB + nt * 16 + l16) * (G_ * D_) + j * 32 + hf * 16;
        bk.q[0] = *(const uint4*)(kp);
        bk.q[1] = *(const uint4*)(kp + 8);
        sv = __builtin_amdgcn_wmma_f32_16x16x32_bf16(
            false, qa[j].bf, false, bk.bf, (short)0, sv, false, false);
      }
      sfr[nt] = sv;
    }

    // ---- causal mask + online softmax ----
    #pragma unroll
    for (int r = 0; r < 8; ++r) {
      const int row = qbase + r + hf * 8;
      float mx = -__builtin_inff();
      #pragma unroll
      for (int nt = 0; nt < 2; ++nt) {
        const int col = kbk * NB + nt * 16 + l16;
        float val = sfr[nt][r];
        if (col > row) val = NEGV;
        sfr[nt][r] = val;
        mx = fmaxf(mx, val);
      }
      #pragma unroll
      for (int off = 8; off > 0; off >>= 1) mx = fmaxf(mx, __shfl_xor(mx, off, 32));
      const float mnew = fmaxf(mrow[r], mx);
      const float scl  = __expf(mrow[r] - mnew);
      float rsum = 0.0f;
      #pragma unroll
      for (int nt = 0; nt < 2; ++nt) {
        const float pv = __expf(sfr[nt][r] - mnew);
        sfr[nt][r] = pv;
        rsum += pv;
      }
      #pragma unroll
      for (int off = 8; off > 0; off >>= 1) rsum += __shfl_xor(rsum, off, 32);
      lrow[r] = lrow[r] * scl + rsum;
      mrow[r] = mnew;
      #pragma unroll
      for (int dt = 0; dt < 4; ++dt) o[dt][r] *= scl;
    }

    // ---- P (C layout) -> per-wave LDS -> A layout (in-wave, no barrier) ----
    #pragma unroll
    for (int nt = 0; nt < 2; ++nt)
      #pragma unroll
      for (int r = 0; r < 8; ++r)
        sP16[wave][(r + hf * 8) * (LPA * 2) + nt * 16 + l16] = bf16_1(sfr[nt][r]);

    Frag pa;
    {
      const uint32_t* rp = (const uint32_t*)&sP16[wave][l16 * (LPA * 2)];
      pa.q[0] = *(const uint4*)(rp + hf * 8);
      pa.q[1] = *(const uint4*)(rp + hf * 8 + 4);
    }

    // ---- O += P V : 4 d-tiles, B-frags straight from V^T rows ----
    #pragma unroll
    for (int dt = 0; dt < 4; ++dt) {
      Frag bv;
      const uint16_t* vp =
          vbase + (size_t)(dt * 16 + l16) * S_ + kbk * NB + hf * 16;
      bv.q[0] = *(const uint4*)(vp);
      bv.q[1] = *(const uint4*)(vp + 8);
      o[dt] = __builtin_amdgcn_wmma_f32_16x16x32_bf16(
          false, pa.bf, false, bv.bf, (short)0, o[dt], false, false);
    }
  }

  // ---- normalize and write O (bf16, feeds output GEMM) ----
  #pragma unroll
  for (int r = 0; r < 8; ++r) {
    const float inv = 1.0f / lrow[r];
    const int row = qbase + r + hf * 8;
    #pragma unroll
    for (int dt = 0; dt < 4; ++dt) {
      const int d = dt * 16 + l16;
      ob[((size_t)(b * T_ + row) * H_ + h) * D_ + d] = bf16_1(o[dt][r] * inv);
    }
  }
}

// =====================================================================
extern "C" void kernel_launch(void* const* d_in, const int* in_sizes, int n_in,
                              void* d_out, int out_size, void* d_ws, size_t ws_size,
                              hipStream_t stream)
{
  (void)in_sizes; (void)n_in; (void)out_size; (void)ws_size;

  const float* x    = (const float*)d_in[0];
  const float* Wqkv = (const float*)d_in[1];
  const float* Wout = (const float*)d_in[2];
  const float* qnw  = (const float*)d_in[3];
  const float* knw  = (const float*)d_in[4];
  const float* cosp = (const float*)d_in[5];
  const float* sinp = (const float*)d_in[6];
  // d_in[7] = mask (unused: causal handled analytically)
  const int* ipos   = (const int*)d_in[10];

  float* y  = (float*)d_out;
  float* ck = y  + (size_t)B_ * T_ * E_;
  float* cv = ck + (size_t)B_ * S_ * G_ * D_;

  uint8_t* ws = (uint8_t*)d_ws;
  const size_t MB = 1024u * 1024u;
  uint16_t* xb  = (uint16_t*)(ws);                 // 16 MB  [B*T, E] bf16
  uint16_t* ob  = xb;                              // alias: reused after GEMM1
  uint16_t* wqb = (uint16_t*)(ws + 16  * MB);      // 12 MB  [3072, E] bf16
  uint16_t* wob = (uint16_t*)(ws + 28  * MB);      //  8 MB  [E, 2048] bf16
  float*    qkv = (float*)   (ws + 36  * MB);      // 48 MB  [B*T, 3072] f32
  uint16_t* qb  = (uint16_t*)(ws + 84  * MB);      // 16 MB  [B,T,H,D] bf16
  uint16_t* kbs = (uint16_t*)(ws + 100 * MB);      //  4 MB  [B,S,G,D] bf16
  uint16_t* vtb = (uint16_t*)(ws + 104 * MB);      //  4 MB  [B,G,D,S] bf16

  // 0) one-time fp32 -> bf16 conversion of GEMM operands
  convert_f32_bf16<<<(B_*T_*E_/2 + 255)/256, 256, 0, stream>>>(x, (uint32_t*)xb, B_*T_*E_/2);
  convert_f32_bf16<<<(QKV_F*E_/2 + 255)/256, 256, 0, stream>>>(Wqkv, (uint32_t*)wqb, QKV_F*E_/2);
  convert_f32_bf16<<<(E_*E_/2 + 255)/256, 256, 0, stream>>>(Wout, (uint32_t*)wob, E_*E_/2);

  // 1) QKV projection
  {
    dim3 g(QKV_F / BN, (B_ * T_) / BM);
    gemm_bf16_wmma<<<g, 256, 0, stream>>>(xb, wqb, qkv, B_ * T_, QKV_F, E_);
  }
  // 2) RMS-norm + RoPE + cache scatter (+ bf16 shadows for attention)
  norm_rope_cache<<<B_ * T_, 256, 0, stream>>>(qkv, qnw, knw, cosp, sinp, ipos,
                                               qb, ck, cv, kbs, vtb);
  // 3) causal flash attention
  attention_kernel<<<B_ * H_ * (T_ / 64), 128, 0, stream>>>(qb, kbs, vtb, ob);
  // 4) output projection
  {
    dim3 g(E_ / BN, (B_ * T_) / BM);
    gemm_bf16_wmma<<<g, 256, 0, stream>>>(ob, wob, y, B_ * T_, E_, E_);
  }
}
